// GraphStateRepresentation_40303973106315
// MI455X (gfx1250) — matrix operand order repaired and checked by reading, not compiled, not run on hardware
//
#include <hip/hip_runtime.h>
#include <hip/hip_bf16.h>

// ---------------------------------------------------------------------------
// GraphStateRepresentation for MI455X (gfx1250, wave32, WMMA).
//
// All dense linear algebra runs through v_wmma_f32_16x16x32_f16 (f16 inputs,
// f32 accumulate). Per-edge DxD matvecs are refactored into a typed
// scatter-add (10 edge types) followed by 10 dense WMMA GEMMs.
// The reference's NSTEPS loop is idempotent (source features never change),
// so one message pass + one GRU per timestep reproduces it exactly.
//
// All GEMM A-side activation buffers are zero-padded to K%32==0 so fragment
// loads are unconditional 16B vector loads (global_load_b128) with no
// K-bounds predication.
// ---------------------------------------------------------------------------

typedef __attribute__((ext_vector_type(16))) _Float16 v16h;
typedef __attribute__((ext_vector_type(8)))  float    v8f;
typedef __attribute__((ext_vector_type(4)))  float    v4f;

// Problem constants
constexpr int BS = 4, T = 6, N = 400, E = 3000;
constexpr int D = 100, NS = 30, NET = 10;
constexpr int MROWS = BS * T * N;     // 9600
constexpr int NNODE = BS * N;         // 1600
constexpr int ETOT  = BS * E;         // 12000
constexpr int DP    = 128;            // padded D
constexpr int CATP  = 224;            // padded 2D
constexpr int NSP   = 32;             // padded NS

// Workspace layout (bytes). 256B aligned regions.
constexpr size_t ALGN(size_t x) { return (x + 255) & ~size_t(255); }
constexpr size_t OFF_W_ST   = 0;                                    // 112x32   f16
constexpr size_t OFF_W_C1   = ALGN(OFF_W_ST   + 112*32*2);          // 112x224  f16
constexpr size_t OFF_W_C2   = ALGN(OFF_W_C1   + 112*224*2);         // 112x128  f16
constexpr size_t OFF_W_OWIH = ALGN(OFF_W_C2   + 112*128*2);         // 304x128  f16
constexpr size_t OFF_W_OWHH = ALGN(OFF_W_OWIH + 304*128*2);
constexpr size_t OFF_W_GWIH = ALGN(OFF_W_OWHH + 304*128*2);
constexpr size_t OFF_W_GWHH = ALGN(OFF_W_GWIH + 304*128*2);
constexpr size_t OFF_W_EDGE = ALGN(OFF_W_GWHH + 304*128*2);         // 10x112x128 f16
constexpr size_t OFF_SPAD   = ALGN(OFF_W_EDGE + NET*112*128*2);     // 9600x32  f32
constexpr size_t OFF_CAT    = ALGN(OFF_SPAD + (size_t)MROWS*NSP*4); // 9600x224 f32 (reused: x-tmp, S[10][1600][128])
constexpr size_t OFF_X1     = ALGN(OFF_CAT + (size_t)MROWS*CATP*4); // 9600x128 f32 (reused: g300a+g300b)
constexpr size_t OFF_XR     = ALGN(OFF_X1  + (size_t)MROWS*DP*4);   // 6x1600x128 f32
constexpr size_t OFF_H      = ALGN(OFF_XR  + (size_t)T*NNODE*DP*4); // 1600x128 f32
constexpr size_t OFF_A      = ALGN(OFF_H   + (size_t)NNODE*DP*4);   // 1600x128 f32
constexpr size_t OFF_END    = OFF_A + (size_t)NNODE*DP*4;

__device__ __forceinline__ float sigmf(float x) { return 1.0f / (1.0f + expf(-x)); }

// ---------------------------------------------------------------------------
// Weight conversion: f32 [rows,cols] -> zero-padded f16 [rowsPad, kpad]
// ---------------------------------------------------------------------------
__global__ void cvt_pad(const float* __restrict__ src, _Float16* __restrict__ dst,
                        int rows, int cols, int rowsPad, int kpad) {
  int i = blockIdx.x * blockDim.x + threadIdx.x;
  int total = rowsPad * kpad;
  if (i >= total) return;
  int r = i / kpad, c = i % kpad;
  dst[i] = (r < rows && c < cols) ? (_Float16)src[(size_t)r * cols + c] : (_Float16)0.0f;
}

__global__ void cvt_edge(const float* __restrict__ src, _Float16* __restrict__ dst) {
  int i = blockIdx.x * blockDim.x + threadIdx.x;
  if (i >= NET * 112 * 128) return;
  int c = i % 128, r = (i / 128) % 112, et = i / (112 * 128);
  dst[i] = (r < D && c < D) ? (_Float16)src[(size_t)et * D * D + r * D + c]
                            : (_Float16)0.0f;
}

__global__ void pad_states(const float* __restrict__ src, float* __restrict__ dst) {
  int i = blockIdx.x * blockDim.x + threadIdx.x;   // MROWS*NSP
  if (i >= MROWS * NSP) return;
  int r = i / NSP, c = i % NSP;
  dst[i] = (c < NS) ? src[(size_t)r * NS + c] : 0.0f;
}

__global__ void zero_f32(float* __restrict__ p, int n) {
  int i = blockIdx.x * blockDim.x + threadIdx.x;
  if (i < n) p[i] = 0.0f;
}

// ---------------------------------------------------------------------------
// A-fragment loader: A 16x32 frags in contiguous k order (lane half h: k = 16h..16h+15); arow stride = NK*32,
// zero-padded, so loads are unconditional 16B vector loads.
// ---------------------------------------------------------------------------
template <int NK>
__device__ __forceinline__ void load_a_frags(const float* __restrict__ arow,
                                             int hi, v16h af[NK]) {
#pragma unroll
  for (int kc = 0; kc < NK; ++kc) {
    const float* p0 = arow + kc * 32 + hi * 16;    // 16B aligned
    v4f a0 = *reinterpret_cast<const v4f*>(p0);
    v4f a1 = *reinterpret_cast<const v4f*>(p0 + 4);
    v4f a2 = *reinterpret_cast<const v4f*>(p0 + 8);
    v4f a3 = *reinterpret_cast<const v4f*>(p0 + 12);
    v16h f;
#pragma unroll
    for (int j = 0; j < 4; ++j) {
      f[j]      = (_Float16)a0[j];
      f[4 + j]  = (_Float16)a1[j];
      f[8 + j]  = (_Float16)a2[j];
      f[12 + j] = (_Float16)a3[j];
    }
    af[kc] = f;
  }
}

// ---------------------------------------------------------------------------
// Generic WMMA GEMM: C[M, Nout] = act(A[M, NK*32] @ W^T + bias)
//   A: f32 row-major, stride NK*32, zero-padded.  W: padded f16 [NTILES*16][NK*32].
//   One wave per 16-row C strip (all NTILES tiles); wave holds all A frags, streams B across N.
//   grid.x = M/64, block = 128 (4 waves).
// ---------------------------------------------------------------------------
template <int NK, int NTILES>
__global__ void __launch_bounds__(128)
gemm_wmma(const float* __restrict__ A, const _Float16* __restrict__ W,
          const float* __restrict__ bias, float* __restrict__ C,
          int ldc, int Nout, int relu) {
  const int lane  = threadIdx.x & 31;
  const int wave  = threadIdx.x >> 5;
  const int mbase = (blockIdx.x * 4 + wave) * 16;
  const int hi    = lane >> 4;          // half-wave select
  const int m     = mbase + (lane & 15);
  constexpr int KP = NK * 32;

  v16h af[NK];
  load_a_frags<NK>(A + (size_t)m * KP, hi, af);

  const int ncl = lane & 15;
#pragma unroll 1
  for (int nt = 0; nt < NTILES; ++nt) {
    v8f acc = {};
    // B fragment: lane owns column n, 16 contiguous K values (32B aligned).
    const _Float16* wb = W + (size_t)(nt * 16 + ncl) * KP + hi * 16;
#pragma unroll
    for (int kc = 0; kc < NK; ++kc) {
      v16h bf = *reinterpret_cast<const v16h*>(wb + kc * 32);
      acc = __builtin_amdgcn_wmma_f32_16x16x32_f16(false, af[kc], false, bf,
                                                   (short)0, acc, false, false);
    }
    int ncol = nt * 16 + ncl;
    if (ncol < Nout) {
      float bv = bias ? bias[ncol] : 0.0f;
#pragma unroll
      for (int r = 0; r < 8; ++r) {
        int mr = mbase + hi * 8 + r;    // f32 C/D layout (ISA 7.12.2)
        float v = acc[r] + bv;
        if (relu) v = fmaxf(v, 0.0f);
        C[(size_t)mr * ldc + ncol] = v;
      }
    }
  }
}

// ---------------------------------------------------------------------------
// Edge-type fused GEMM: a[1600,128p] = sum_k S[k] @ Wedge[k]^T
//   S: f32 [NET][1600][128p], Wedge: f16 [NET][112][128].
//   grid.x = 25, block = 128. 280 wmma per wave.
// ---------------------------------------------------------------------------
__global__ void __launch_bounds__(128)
gemm_edge_types(const float* __restrict__ S, const _Float16* __restrict__ Wedge,
                float* __restrict__ Aout) {
  const int lane  = threadIdx.x & 31;
  const int wave  = threadIdx.x >> 5;
  const int mbase = (blockIdx.x * 4 + wave) * 16;
  const int hi    = lane >> 4;
  const int m     = mbase + (lane & 15);
  const int ncl   = lane & 15;

  v8f acc[7];
#pragma unroll
  for (int nt = 0; nt < 7; ++nt) acc[nt] = (v8f){};

#pragma unroll 1
  for (int et = 0; et < NET; ++et) {
    v16h af[4];
    load_a_frags<4>(S + ((size_t)et * NNODE + m) * DP, hi, af);
    const _Float16* wbase = Wedge + (size_t)et * 112 * 128;
#pragma unroll
    for (int nt = 0; nt < 7; ++nt) {
      const _Float16* wb = wbase + (size_t)(nt * 16 + ncl) * 128 + hi * 16;
#pragma unroll
      for (int kc = 0; kc < 4; ++kc) {
        v16h bf = *reinterpret_cast<const v16h*>(wb + kc * 32);
        acc[nt] = __builtin_amdgcn_wmma_f32_16x16x32_f16(false, af[kc], false, bf,
                                                         (short)0, acc[nt], false, false);
      }
    }
  }
#pragma unroll
  for (int nt = 0; nt < 7; ++nt) {
    int ncol = nt * 16 + ncl;
    if (ncol < D) {
#pragma unroll
      for (int r = 0; r < 8; ++r) {
        int mr = mbase + hi * 8 + r;
        Aout[(size_t)mr * DP + ncol] = acc[nt][r];
      }
    }
  }
}

// ---------------------------------------------------------------------------
// Small elementwise / data-movement kernels
// ---------------------------------------------------------------------------
__global__ void build_cat_emb(const int* __restrict__ cls,
                              const float* __restrict__ obj_emb,
                              float* __restrict__ cat) {
  int i = blockIdx.x * blockDim.x + threadIdx.x;    // MROWS*D
  if (i >= MROWS * D) return;
  int d = i % D, row = i / D;
  cat[(size_t)row * CATP + d] = obj_emb[(size_t)cls[row] * D + d];
}

__global__ void permute_x(const float* __restrict__ x, float* __restrict__ xr) {
  int i = blockIdx.x * blockDim.x + threadIdx.x;    // MROWS*D
  if (i >= MROWS * D) return;
  int d = i % D, row = i / D;
  int node = row % N, t = (row / N) % T, b = row / (N * T);
  xr[((size_t)(t * BS + b) * N + node) * DP + d] = x[i];
}

__global__ void gru_outer(const float* __restrict__ gi, const float* __restrict__ gh,
                          float* __restrict__ h) {
  int i = blockIdx.x * blockDim.x + threadIdx.x;    // NNODE*D
  if (i >= NNODE * D) return;
  int d = i % D, m = i / D;
  const float* gib = gi + (size_t)m * 300;
  const float* ghb = gh + (size_t)m * 300;
  float r  = sigmf(gib[d] + ghb[d]);
  float z  = sigmf(gib[100 + d] + ghb[100 + d]);
  float nn = tanhf(gib[200 + d] + r * ghb[200 + d]);
  size_t hidx = (size_t)m * DP + d;
  h[hidx] = (1.0f - z) * nn + z * h[hidx];
}

__global__ void scatter_typed(const float* __restrict__ h, const int* __restrict__ edges,
                              const int* __restrict__ etypes, const float* __restrict__ mask,
                              int t, float* __restrict__ S) {
  int gw = (blockIdx.x * blockDim.x + threadIdx.x) >> 5;   // wave == edge
  int lane = threadIdx.x & 31;
  if (gw >= ETOT) return;
  int b = gw / E, e = gw % E;
  size_t eb = ((size_t)(b * T + t) * E + e);
  int src = edges[eb * 2 + 0] + b * N;
  int dst = edges[eb * 2 + 1] + b * N;
  int et  = etypes[eb];
  float mk = mask[eb];
  if (mk == 0.0f) return;
  const float* hs = h + (size_t)src * DP;
  float* sp = S + ((size_t)et * NNODE + dst) * DP;
  for (int d = lane; d < D; d += 32) atomicAdd(sp + d, mk * hs[d]);
}

__global__ void gru_inner_out(const float* __restrict__ gi, const float* __restrict__ gh,
                              const float* __restrict__ h, const float* __restrict__ vis,
                              int t, float* __restrict__ outN) {
  int i = blockIdx.x * blockDim.x + threadIdx.x;    // NNODE*D
  if (i >= NNODE * D) return;
  int d = i % D, m = i / D;
  int b = m / N, node = m % N;
  const float* gib = gi + (size_t)m * 300;
  const float* ghb = gh + (size_t)m * 300;
  float r  = sigmf(gib[d] + ghb[d]);
  float z  = sigmf(gib[100 + d] + ghb[100 + d]);
  float nn = tanhf(gib[200 + d] + r * ghb[200 + d]);
  float feat = (1.0f - z) * nn + z * h[(size_t)m * DP + d];
  size_t btn = (size_t)(b * T + t) * N + node;
  outN[btn * D + d] = feat * vis[btn];
}

__global__ void global_reduce(const float* __restrict__ nodeOut,
                              const float* __restrict__ vis, float* __restrict__ gout) {
  int bt = blockIdx.x;                 // b*T + t
  int d = threadIdx.x;
  if (d >= D) return;
  const float* base = nodeOut + (size_t)bt * N * D;
  const float* vp = vis + (size_t)bt * N;
  float s = 0.0f, vsum = 0.0f;
  for (int nIdx = 0; nIdx < N; ++nIdx) {
    float v = vp[nIdx];
    s += base[(size_t)nIdx * D + d] * v;   // node_reprs already has one vis factor
    vsum += v;
  }
  gout[(size_t)bt * D + d] = s / (vsum + 1e-9f);
}

// ---------------------------------------------------------------------------
extern "C" void kernel_launch(void* const* d_in, const int* in_sizes, int n_in,
                              void* d_out, int out_size, void* d_ws, size_t ws_size,
                              hipStream_t stream) {
  (void)in_sizes; (void)n_in; (void)out_size; (void)ws_size;
  const int*   class_names = (const int*)d_in[0];
  const float* states      = (const float*)d_in[1];
  const int*   edges       = (const int*)d_in[2];
  const int*   edge_types  = (const int*)d_in[3];
  const float* visibility  = (const float*)d_in[4];
  const float* mask_edges  = (const float*)d_in[5];
  const float* obj_emb     = (const float*)d_in[6];
  const float* state_W     = (const float*)d_in[7];
  const float* state_b     = (const float*)d_in[8];
  const float* c1_W        = (const float*)d_in[9];
  const float* c1_b        = (const float*)d_in[10];
  const float* c2_W        = (const float*)d_in[11];
  const float* c2_b        = (const float*)d_in[12];
  const float* edge_embed  = (const float*)d_in[13];
  const float* g_wih       = (const float*)d_in[14];
  const float* g_whh       = (const float*)d_in[15];
  const float* g_bih       = (const float*)d_in[16];
  const float* g_bhh       = (const float*)d_in[17];
  const float* o_wih       = (const float*)d_in[18];
  const float* o_whh       = (const float*)d_in[19];

  char* ws = (char*)d_ws;
  _Float16* w_st   = (_Float16*)(ws + OFF_W_ST);
  _Float16* w_c1   = (_Float16*)(ws + OFF_W_C1);
  _Float16* w_c2   = (_Float16*)(ws + OFF_W_C2);
  _Float16* w_owih = (_Float16*)(ws + OFF_W_OWIH);
  _Float16* w_owhh = (_Float16*)(ws + OFF_W_OWHH);
  _Float16* w_gwih = (_Float16*)(ws + OFF_W_GWIH);
  _Float16* w_gwhh = (_Float16*)(ws + OFF_W_GWHH);
  _Float16* w_edge = (_Float16*)(ws + OFF_W_EDGE);
  float* spad  = (float*)(ws + OFF_SPAD);   // states padded [9600][32]
  float* cat   = (float*)(ws + OFF_CAT);    // [9600][224]; reused: x-tmp, S
  float* x1    = (float*)(ws + OFF_X1);     // [9600][128]; reused: g300a/g300b
  float* xr    = (float*)(ws + OFF_XR);     // [6][1600][128]
  float* h     = (float*)(ws + OFF_H);      // [1600][128]
  float* abuf  = (float*)(ws + OFF_A);      // [1600][128]
  float* xtmp  = cat;                       // MLP output before permute (dense 100)
  float* Sbuf  = cat;                       // typed scatter buffer during t-loop
  float* g300a = x1;                        // 1600x300
  float* g300b = x1 + (size_t)NNODE * 300;  // 1600x300

  float* out_nodes  = (float*)d_out;                          // [4,6,400,100]
  float* out_global = (float*)d_out + (size_t)BS * T * N * D; // [4,6,100]

  const int BT = 256;
  auto blocks = [](int n, int bs) { return (n + bs - 1) / bs; };

  // 0) Zero the activation region once (establishes all pad columns = 0).
  {
    int nz = (int)((OFF_END - OFF_CAT) / 4);
    zero_f32<<<blocks(nz, BT), BT, 0, stream>>>((float*)(ws + OFF_CAT), nz);
  }

  // 1) Convert weights to padded f16; pad states.
  cvt_pad<<<blocks(112 * 32, BT), BT, 0, stream>>>(state_W, w_st, D, NS, 112, 32);
  cvt_pad<<<blocks(112 * 224, BT), BT, 0, stream>>>(c1_W, w_c1, D, 2 * D, 112, 224);
  cvt_pad<<<blocks(112 * 128, BT), BT, 0, stream>>>(c2_W, w_c2, D, D, 112, 128);
  cvt_pad<<<blocks(304 * 128, BT), BT, 0, stream>>>(o_wih, w_owih, 300, D, 304, 128);
  cvt_pad<<<blocks(304 * 128, BT), BT, 0, stream>>>(o_whh, w_owhh, 300, D, 304, 128);
  cvt_pad<<<blocks(304 * 128, BT), BT, 0, stream>>>(g_wih, w_gwih, 300, D, 304, 128);
  cvt_pad<<<blocks(304 * 128, BT), BT, 0, stream>>>(g_whh, w_gwhh, 300, D, 304, 128);
  cvt_edge<<<blocks(NET * 112 * 128, BT), BT, 0, stream>>>(edge_embed, w_edge);
  pad_states<<<blocks(MROWS * NSP, BT), BT, 0, stream>>>(states, spad);

  // 2) Input MLP: cat = [emb | states@stW^T+b]; x1 = relu(cat@c1^T+b1); xtmp = x1@c2^T+b2
  build_cat_emb<<<blocks(MROWS * D, BT), BT, 0, stream>>>(class_names, obj_emb, cat);
  gemm_wmma<1, 7><<<MROWS / 64, 128, 0, stream>>>(spad, w_st, state_b, cat + D, CATP, D, 0);
  gemm_wmma<7, 7><<<MROWS / 64, 128, 0, stream>>>(cat, w_c1, c1_b, x1, DP, D, 1);
  gemm_wmma<4, 7><<<MROWS / 64, 128, 0, stream>>>(x1, w_c2, c2_b, xtmp, D, D, 0);
  permute_x<<<blocks(MROWS * D, BT), BT, 0, stream>>>(xtmp, xr);
  zero_f32<<<blocks(NNODE * DP, BT), BT, 0, stream>>>(h, NNODE * DP);

  // 3) Temporal loop
  for (int t = 0; t < T; ++t) {
    const float* xt = xr + (size_t)t * NNODE * DP;
    // outer GRU (no bias): h = gru(x_t, h)
    gemm_wmma<4, 19><<<NNODE / 64, 128, 0, stream>>>(xt, w_owih, nullptr, g300a, 300, 300, 0);
    gemm_wmma<4, 19><<<NNODE / 64, 128, 0, stream>>>(h, w_owhh, nullptr, g300b, 300, 300, 0);
    gru_outer<<<blocks(NNODE * D, BT), BT, 0, stream>>>(g300a, g300b, h);

    // typed message scatter: S[type,dst,:] += mask * h[src,:]
    zero_f32<<<blocks(NET * NNODE * DP, BT), BT, 0, stream>>>(Sbuf, NET * NNODE * DP);
    scatter_typed<<<ETOT / 8, BT, 0, stream>>>(h, edges, edge_types, mask_edges, t, Sbuf);

    // a = sum_k S[k] @ W_k^T  (10 dense WMMA GEMMs fused)
    gemm_edge_types<<<NNODE / 64, 128, 0, stream>>>(Sbuf, w_edge, abuf);

    // inner GRU (with bias) + visibility, writes node_reprs[:, t]
    gemm_wmma<4, 19><<<NNODE / 64, 128, 0, stream>>>(abuf, w_gwih, g_bih, g300a, 300, 300, 0);
    gemm_wmma<4, 19><<<NNODE / 64, 128, 0, stream>>>(h, w_gwhh, g_bhh, g300b, 300, 300, 0);
    gru_inner_out<<<blocks(NNODE * D, BT), BT, 0, stream>>>(g300a, g300b, h,
                                                            visibility, t, out_nodes);
  }

  // 4) global_repr = sum_n(node_reprs * vis) / (sum_n vis + 1e-9)
  global_reduce<<<BS * T, 128, 0, stream>>>(out_nodes, visibility, out_global);
}
